// PointMambaEncoder_35407710389124
// MI455X (gfx1250) — matrix operand order repaired and checked by reading, not compiled, not run on hardware
//
#include <hip/hip_runtime.h>
#include <stdint.h>

typedef __attribute__((ext_vector_type(16))) __bf16 v16bf;
typedef __attribute__((ext_vector_type(8)))  float  v8f;

#define BN_INV 0.99999500003f    // 1/sqrt(1+1e-5)
#define PI_F 3.14159265358979323846f

__device__ __forceinline__ unsigned short f2bf(float x){
  unsigned u = __float_as_uint(x);
  return (unsigned short)((u + 0x7FFFu + ((u >> 16) & 1u)) >> 16);   // RNE
}

// ---------------- weight / BN prep ----------------
// src OIDHW f32 -> dst [k][co][ci] bf16, k = kd*9+kh*3+kw
template<int CIN, int COUT>
__global__ void prep_w_kernel(const float* __restrict__ w, unsigned short* __restrict__ wp){
  int i = blockIdx.x * 256 + threadIdx.x;
  const int total = 27 * CIN * COUT;
  if (i >= total) return;
  int ci = i % CIN;
  int co = (i / CIN) % COUT;
  int k  = i / (CIN * COUT);
  wp[i] = f2bf(w[(co * CIN + ci) * 27 + k]);
}

__global__ void prep_bn_kernel(const float* __restrict__ bconv, const float* __restrict__ g,
                               const float* __restrict__ be, float* __restrict__ sc,
                               float* __restrict__ bi, int C){
  int i = blockIdx.x * 256 + threadIdx.x;
  if (i < C){ float s = g[i] * BN_INV; sc[i] = s; bi[i] = bconv[i] * s + be[i]; }
}

__global__ void copy_xyz_kernel(const float* __restrict__ xyz, float* __restrict__ out){
  int i = blockIdx.x * 256 + threadIdx.x;
  if (i < 3072) out[i] = xyz[i];
}

// ---------------- furthest point sampling (N=1024, M=256) ----------------
__global__ __launch_bounds__(1024) void fps_kernel(const float* __restrict__ xyz,
                                                   float* __restrict__ kpts){
  const int tid = threadIdx.x;
  __shared__ float red_v[32];
  __shared__ int   red_i[32];
  __shared__ int   s_last;
  const float px = xyz[tid*3+0], py = xyz[tid*3+1], pz = xyz[tid*3+2];
  float d = 1e10f;
  int last = 0;
  for (int it = 0; it < 256; ++it){
    if (tid == 0){
      kpts[it*3+0] = xyz[last*3+0];
      kpts[it*3+1] = xyz[last*3+1];
      kpts[it*3+2] = xyz[last*3+2];
    }
    const float qx = xyz[last*3+0], qy = xyz[last*3+1], qz = xyz[last*3+2];
    const float dx = px-qx, dy = py-qy, dz = pz-qz;
    d = fminf(d, dx*dx + dy*dy + dz*dz);
    float v = d; int idx = tid;
    for (int off = 16; off >= 1; off >>= 1){           // argmax, first-index tie-break
      float ov = __shfl_xor(v, off);
      int   oi = __shfl_xor(idx, off);
      if (ov > v || (ov == v && oi < idx)) { v = ov; idx = oi; }
    }
    if ((tid & 31) == 0){ red_v[tid >> 5] = v; red_i[tid >> 5] = idx; }
    __syncthreads();
    if (tid < 32){
      float v2 = red_v[tid]; int i2 = red_i[tid];
      for (int off = 16; off >= 1; off >>= 1){
        float ov = __shfl_xor(v2, off);
        int   oi = __shfl_xor(i2, off);
        if (ov > v2 || (ov == v2 && oi < i2)) { v2 = ov; i2 = oi; }
      }
      if (tid == 0) s_last = i2;
    }
    __syncthreads();
    last = s_last;
  }
}

// ---------------- spherical binning + xyz_raising + BN + ReLU -> bf16 ----------------
// one block per keypoint; analytic per-dimension nearest-bin (regular product grid)
__global__ __launch_bounds__(256) void bin_raise_kernel(
    const float* __restrict__ xyz, const float* __restrict__ kpts,
    const float* __restrict__ w_raise, const float* __restrict__ b_raise,
    const float* __restrict__ g0, const float* __restrict__ be0,
    unsigned short* __restrict__ x0)
{
  const int m   = blockIdx.x;
  const int tid = threadIdx.x;
  __shared__ float s_rel[1024*3];
  __shared__ int   s_idx[1024];
  const float kx = kpts[m*3+0], ky = kpts[m*3+1], kz = kpts[m*3+2];
  for (int n = tid; n < 1024; n += 256){
    float rx = xyz[n*3+0]-kx, ry = xyz[n*3+1]-ky, rz = xyz[n*3+2]-kz;
    float rho = sqrtf(rx*rx + ry*ry + rz*rz);
    float th  = acosf(rz / (rho + 1e-8f));
    float ph  = atan2f(ry, rx);
    int kr = (int)floorf(rho * 16.0f);               // (rho/0.5)*8
    kr = min(max(kr, 0), 7);
    int kt = (int)floorf(th * (8.0f / PI_F));        kt = min(max(kt, 0), 7);
    int kp = (int)floorf(ph * (4.0f / PI_F) + 4.0f); kp = min(max(kp, 0), 7);
    s_idx[n] = kr*64 + kt*8 + kp;
    s_rel[n*3+0] = rx; s_rel[n*3+1] = ry; s_rel[n*3+2] = rz;
  }
  __syncthreads();
  // deterministic: each thread owns bins 2*tid, 2*tid+1, scans points in order
  const int b0 = tid * 2;
  float sx0=0,sy0=0,sz0=0,c0v=0, sx1=0,sy1=0,sz1=0,c1v=0;
  for (int n = 0; n < 1024; ++n){
    int ix = s_idx[n];
    if (ix == b0)      { sx0+=s_rel[n*3]; sy0+=s_rel[n*3+1]; sz0+=s_rel[n*3+2]; c0v+=1.f; }
    else if (ix == b0+1){ sx1+=s_rel[n*3]; sy1+=s_rel[n*3+1]; sz1+=s_rel[n*3+2]; c1v+=1.f; }
  }
  for (int q = 0; q < 2; ++q){
    const int k = b0 + q;
    float c  = fmaxf(q ? c1v : c0v, 1.0f);
    float fx = (q ? sx1 : sx0) / c;
    float fy = (q ? sy1 : sy0) / c;
    float fz = (q ? sz1 : sz0) / c;
    unsigned short* dst = x0 + ((size_t)m * 512 + k) * 32;
    for (int o = 0; o < 32; ++o){
      float v = w_raise[o*3+0]*fx + w_raise[o*3+1]*fy + w_raise[o*3+2]*fz + b_raise[o];
      v = v * (g0[o] * BN_INV) + be0[o];
      dst[o] = f2bf(fmaxf(v, 0.0f));
    }
  }
}

// ---------------- Conv3d 3x3x3 SAME + BN + ReLU as implicit-GEMM WMMA ----------------
// one block per batch element (keypoint). 512 threads = 16 wave32.
// Each wave-task owns 2 co-tiles x 4 spatial-tiles: every B fragment feeds 2 WMMAs;
// out-of-bounds taps read a zeroed LDS pad (address select only, no data cndmask).
template<int CIN, int COUT, bool DO_MAX>
__global__ __launch_bounds__(512) void conv3d_wmma_kernel(
    const unsigned short* __restrict__ xin, const unsigned short* __restrict__ wprep,
    const float* __restrict__ sc, const float* __restrict__ bi,
    unsigned short* __restrict__ xout, float* __restrict__ desc)
{
  const int b    = blockIdx.x;
  const int tid  = threadIdx.x;
  const int wave = tid >> 5;
  const int lane = tid & 31;
  const int half = lane >> 4;
  const int nn   = lane & 15;

  constexpr int PAD = 512 * CIN;                   // zero-pad region base (ushort idx)
  __shared__ unsigned short lds_x[512 * CIN + 16]; // 8x8x8 x CIN bf16 volume + 32B zeros
  __shared__ int lds_max[DO_MAX ? COUT : 1];

  { // stage the whole input volume into LDS, b128 vectors
    const uint4* src = (const uint4*)(xin + (size_t)b * 512 * CIN);
    uint4* dst = (uint4*)lds_x;
    const int VE = (512 * CIN) / 8;
    for (int i = tid; i < VE; i += 512) dst[i] = src[i];
  }
  if (tid < 16) lds_x[PAD + tid] = 0;
  if (DO_MAX){ for (int i = tid; i < COUT; i += 512) lds_max[i] = 0; }
  __syncthreads();

  constexpr int SPT    = 4;               // spatial 16-tiles per wave-task
  constexpr int COP    = 2;               // co-tiles per wave-task (B reuse)
  constexpr int CO_T   = COUT / 16;
  constexpr int SGRP   = 32 / SPT;        // 8 spatial groups
  constexpr int TASKS  = (CO_T / COP) * SGRP;  // 16 / 32 / 64 (all % 16 == 0)
  constexpr int KCH    = CIN / 32;        // 32-wide K chunks per kernel offset
  constexpr int CHUNKS = 27 * KCH;

  for (int task = wave; task < TASKS; task += 16){
    const int tp = task / SGRP;           // co tile-pair
    const int sg = task % SGRP;
    const int s0 = sg * (SPT * 16);
    int sC[SPT], sd[SPT], shh[SPT], sww[SPT];
    for (int j = 0; j < SPT; ++j){
      int s = s0 + j*16 + nn;
      sd[j] = s >> 6; shh[j] = (s >> 3) & 7; sww[j] = s & 7;
      sC[j] = s * CIN;
    }
    v8f acc[COP][SPT] = {};
    for (int ch = 0; ch < CHUNKS; ++ch){
      const int k  = ch / KCH;
      const int c0 = (ch % KCH) * 32;
      const int kd = k / 9 - 1;
      const int kh = (k / 3) % 3 - 1;
      const int kw = k % 3 - 1;
      const int soff = (kd*64 + kh*8 + kw) * CIN + c0 + half*16;  // scalar per chunk
      // A fragments: lane holds co = tile*16+nn ; contiguous K (k = 16*half + s), matching B
      union { v16bf v; uint4 q[2]; } af[COP];
      for (int p = 0; p < COP; ++p){
        const unsigned short* wp =
            wprep + ((size_t)k * COUT + (tp*COP + p)*16 + nn) * CIN + c0 + half*16;
        af[p].q[0] = *(const uint4*)(wp);
        af[p].q[1] = *(const uint4*)(wp + 8);
      }
      for (int j = 0; j < SPT; ++j){
        const int d = sd[j] + kd, h = shh[j] + kh, w = sww[j] + kw;
        const bool ok = ((unsigned)d < 8u) && ((unsigned)h < 8u) && ((unsigned)w < 8u);
        const int base = ok ? (sC[j] + soff) : PAD;   // OOB -> zeroed pad
        union { v16bf v; uint4 q[2]; } bfr;
        bfr.q[0] = *(const uint4*)(lds_x + base);
        bfr.q[1] = *(const uint4*)(lds_x + base + 8);
        for (int p = 0; p < COP; ++p)
          acc[p][j] = __builtin_amdgcn_wmma_f32_16x16x32_bf16(
                        false, af[p].v, false, bfr.v, (short)0, acc[p][j], false, false);
      }
    }
    // epilogue: BN + ReLU, pack bf16 (D layout: lane n, VGPR r -> co = cobase+r)
    for (int p = 0; p < COP; ++p){
      const int cobase = (tp*COP + p)*16 + half*8;
      for (int j = 0; j < SPT; ++j){
        const int s = s0 + j*16 + nn;
        union { unsigned short us[8]; uint4 q; } o;
        for (int r = 0; r < 8; ++r){
          float v = acc[p][j][r] * sc[cobase + r] + bi[cobase + r];
          v = fmaxf(v, 0.0f);
          if (DO_MAX) atomicMax(&lds_max[cobase + r], __float_as_int(v)); // v>=0: bit-monotone
          o.us[r] = f2bf(v);
        }
        if (!DO_MAX)
          *(uint4*)(xout + ((size_t)b * 512 + s) * COUT + cobase) = o.q;
      }
    }
  }
  if (DO_MAX){
    __syncthreads();
    for (int i = tid; i < COUT; i += 512)
      desc[(size_t)b * COUT + i] = __int_as_float(lds_max[i]);
  }
}

// ---------------- P[m][o] = sum_c desc[m][c] * w_proj[o][c] ----------------
__global__ __launch_bounds__(128) void proj_kernel(const float* __restrict__ desc,
                                                   const float* __restrict__ w_proj,
                                                   float* __restrict__ P){
  const int m = blockIdx.x, o = threadIdx.x;
  float s = 0.f;
  for (int c = 0; c < 256; ++c) s += desc[m*256 + c] * w_proj[o*256 + c];
  P[m*128 + o] = s;
}

// ---------------- IDW interpolation of projected descriptors ----------------
__global__ __launch_bounds__(128) void interp_kernel(const float* __restrict__ xyz,
                                                     const float* __restrict__ kpts,
                                                     const float* __restrict__ P,
                                                     float* __restrict__ out){
  const int n = blockIdx.x, tid = threadIdx.x;
  __shared__ float swt[256];
  const float px = xyz[n*3+0], py = xyz[n*3+1], pz = xyz[n*3+2];
  for (int m = tid; m < 256; m += 128){
    float dx = px - kpts[m*3+0], dy = py - kpts[m*3+1], dz = pz - kpts[m*3+2];
    float dist = sqrtf(fmaxf(dx*dx + dy*dy + dz*dz, 0.f));
    swt[m] = 1.0f / (dist + 1e-8f);
  }
  __syncthreads();
  float tot = 0.f;
  for (int m = 0; m < 256; ++m) tot += swt[m];
  float a = 0.f;
  for (int m = 0; m < 256; ++m) a += swt[m] * P[m*128 + tid];
  out[3072 + tid*1024 + n] = a / tot;   // out layout: xyz(3072) then (128,1024)
}

extern "C" void kernel_launch(void* const* d_in, const int* in_sizes, int n_in,
                              void* d_out, int out_size, void* d_ws, size_t ws_size,
                              hipStream_t stream)
{
  (void)in_sizes; (void)n_in; (void)out_size; (void)ws_size;
  const float* xyz     = (const float*)d_in[0];
  const float* w_raise = (const float*)d_in[1];
  const float* b_raise = (const float*)d_in[2];
  const float* g0  = (const float*)d_in[3];
  const float* be0 = (const float*)d_in[4];
  const float* w1  = (const float*)d_in[5];
  const float* b1  = (const float*)d_in[6];
  const float* g1  = (const float*)d_in[7];
  const float* be1 = (const float*)d_in[8];
  const float* w2  = (const float*)d_in[9];
  const float* b2  = (const float*)d_in[10];
  const float* g2  = (const float*)d_in[11];
  const float* be2 = (const float*)d_in[12];
  const float* w3  = (const float*)d_in[13];
  const float* b3  = (const float*)d_in[14];
  const float* g3  = (const float*)d_in[15];
  const float* be3 = (const float*)d_in[16];
  const float* w_proj = (const float*)d_in[17];
  float* out = (float*)d_out;

  char* wsb = (char*)d_ws;
  size_t off = 0;
  auto alloc = [&](size_t bytes)->void* {
    void* p = wsb + off; off = (off + bytes + 255) & ~((size_t)255); return p;
  };
  float*          kpts = (float*)         alloc(256*3*sizeof(float));
  unsigned short* x0   = (unsigned short*)alloc((size_t)256*512*32*2);
  unsigned short* x1   = (unsigned short*)alloc((size_t)256*512*64*2);
  unsigned short* x2   = (unsigned short*)alloc((size_t)256*512*128*2);
  unsigned short* w1p  = (unsigned short*)alloc((size_t)27*64*32*2);
  unsigned short* w2p  = (unsigned short*)alloc((size_t)27*128*64*2);
  unsigned short* w3p  = (unsigned short*)alloc((size_t)27*256*128*2);
  float* sc1 = (float*)alloc(64*4);  float* bi1 = (float*)alloc(64*4);
  float* sc2 = (float*)alloc(128*4); float* bi2 = (float*)alloc(128*4);
  float* sc3 = (float*)alloc(256*4); float* bi3 = (float*)alloc(256*4);
  float* desc = (float*)alloc((size_t)256*256*4);
  float* P    = (float*)alloc((size_t)256*128*4);

  prep_w_kernel<32,64>  <<<(27*32*64   + 255)/256, 256, 0, stream>>>(w1, w1p);
  prep_w_kernel<64,128> <<<(27*64*128  + 255)/256, 256, 0, stream>>>(w2, w2p);
  prep_w_kernel<128,256><<<(27*128*256 + 255)/256, 256, 0, stream>>>(w3, w3p);
  prep_bn_kernel<<<1, 256, 0, stream>>>(b1, g1, be1, sc1, bi1, 64);
  prep_bn_kernel<<<1, 256, 0, stream>>>(b2, g2, be2, sc2, bi2, 128);
  prep_bn_kernel<<<1, 256, 0, stream>>>(b3, g3, be3, sc3, bi3, 256);
  copy_xyz_kernel<<<12, 256, 0, stream>>>(xyz, out);

  fps_kernel<<<1, 1024, 0, stream>>>(xyz, kpts);
  bin_raise_kernel<<<256, 256, 0, stream>>>(xyz, kpts, w_raise, b_raise, g0, be0, x0);

  conv3d_wmma_kernel<32, 64,  false><<<256, 512, 0, stream>>>(x0, w1p, sc1, bi1, x1, nullptr);
  conv3d_wmma_kernel<64, 128, false><<<256, 512, 0, stream>>>(x1, w2p, sc2, bi2, x2, nullptr);
  conv3d_wmma_kernel<128,256, true ><<<256, 512, 0, stream>>>(x2, w3p, sc3, bi3, nullptr, desc);

  proj_kernel<<<256, 128, 0, stream>>>(desc, w_proj, P);
  interp_kernel<<<1024, 128, 0, stream>>>(xyz, kpts, P, out);
}
